// TransformerDecoder_47356309406167
// MI455X (gfx1250) — compile-verified, matrix operand order checked by reading, not run on hardware
//
#include <hip/hip_runtime.h>
#include <hip/hip_bf16.h>

// ---------------------------------------------------------------------------
// Types for CDNA5 WMMA (wave32, v_wmma_f32_16x16x32_bf16)
// ---------------------------------------------------------------------------
typedef __attribute__((ext_vector_type(16))) __bf16 bf16x16;
typedef __attribute__((ext_vector_type(8)))  __bf16 bf16x8;
typedef __attribute__((ext_vector_type(8)))  float  f32x8;

__device__ __forceinline__ f32x8 wmma_bf16(bf16x16 a, bf16x16 b, f32x8 c) {
  return __builtin_amdgcn_wmma_f32_16x16x32_bf16(
      /*neg_a=*/false, a, /*neg_b=*/false, b,
      /*c_mod=*/(short)0, c, /*reuse_a=*/false, /*reuse_b=*/false);
}

// Load a 16-element bf16 A/B fragment from LDS.
// ISA 16-bit layout: lane half 'hi' (0 or 8); element j<8 -> K=hi+j,
// element j>=8 -> K=hi+16+(j-8).  With K contiguous in LDS this is two
// 16-byte ds_load_b128.
__device__ __forceinline__ bf16x16 ld_frag(const __bf16* p) {
  bf16x8 lo = *(const bf16x8*)p;
  bf16x8 hi = *(const bf16x8*)(p + 16);
  return __builtin_shufflevector(lo, hi, 0, 1, 2, 3, 4, 5, 6, 7,
                                 8, 9, 10, 11, 12, 13, 14, 15);
}

__device__ __forceinline__ float hmax16(float v) {
  v = fmaxf(v, __shfl_xor(v, 1, 32));
  v = fmaxf(v, __shfl_xor(v, 2, 32));
  v = fmaxf(v, __shfl_xor(v, 4, 32));
  v = fmaxf(v, __shfl_xor(v, 8, 32));
  return v;
}
__device__ __forceinline__ float hsum16(float v) {
  v += __shfl_xor(v, 1, 32);
  v += __shfl_xor(v, 2, 32);
  v += __shfl_xor(v, 4, 32);
  v += __shfl_xor(v, 8, 32);
  return v;
}

// ---------------------------------------------------------------------------
// Embedding + sinusoidal positions
// ---------------------------------------------------------------------------
__global__ __launch_bounds__(256) void embed_kernel(
    const int* __restrict__ tokens, const float* __restrict__ emb,
    float* __restrict__ h, int S, int D) {
  int s = blockIdx.x;
  int tok = tokens[s];
  const float ln1e4 = 9.210340371976184f;  // log(10000)
  for (int d = threadIdx.x; d < D; d += 256) {
    float de = (float)(d & ~1);
    float ang = (float)s * __expf(-ln1e4 * de / (float)D);
    float p = (d & 1) ? __cosf(ang) : __sinf(ang);
    h[(size_t)s * D + d] = emb[(size_t)tok * D + d] + p;
  }
}

// ---------------------------------------------------------------------------
// Generic GEMM: C[M,N] = epi(A[M,K] @ B[K,N] + bias)
// bf16 WMMA, f32 accumulation. Block tile 128x128, BK=32, 128 threads
// (4 waves, each wave owns a 64x64 sub-tile = 16 accumulators).
// Register double-buffered global->LDS staging: next tile's global loads are
// issued before the current tile's WMMAs so loadcnt waits overlap compute.
// epi: 0 = none, 1 = +bias, 2 = +bias then ReLU
// ---------------------------------------------------------------------------
#define GLDA 40  // LDS inner stride (bf16 elems); 40*2B = 80B, mult of 16

__global__ __launch_bounds__(128) void gemm_kernel(
    const float* __restrict__ A, const float* __restrict__ Bm,
    float* __restrict__ C, const float* __restrict__ bias,
    int M, int N, int K, int epi) {
  __shared__ alignas(16) __bf16 sA[128 * GLDA];   // [row][k]
  __shared__ alignas(16) __bf16 sB[128 * GLDA];   // transposed: [n][k]

  const int tid = threadIdx.x;
  const int lane = tid & 31;
  const int wave = tid >> 5;
  const int wm = (wave & 1) * 64;   // wave's M offset in tile
  const int wn = (wave >> 1) * 64;  // wave's N offset in tile
  const int m0 = blockIdx.y * 128;
  const int n0 = blockIdx.x * 128;
  const int row16 = lane & 15;
  const int hi = (lane >> 4) << 3;  // 0 or 8

  f32x8 acc[4][4];
#pragma unroll
  for (int i = 0; i < 4; ++i)
#pragma unroll
    for (int t = 0; t < 4; ++t) acc[i][t] = (f32x8)0.0f;

  float4 ra[8], rb[8];

  // ---- issue global loads for a k-tile into registers ----
  auto ld_tiles = [&](int k0) {
#pragma unroll
    for (int it = 0; it < 8; ++it) {
      int slot = it * 128 + tid;            // 0..1023 float4 slots (A: 128x32)
      int r = slot >> 3, c4 = (slot & 7) * 4;
      ra[it] = *(const float4*)(A + (size_t)(m0 + r) * K + k0 + c4);
    }
#pragma unroll
    for (int it = 0; it < 8; ++it) {
      int slot = it * 128 + tid;            // B: 32x128
      int r = slot >> 5, c4 = (slot & 31) * 4;
      rb[it] = *(const float4*)(Bm + (size_t)(k0 + r) * N + n0 + c4);
    }
  };

  // ---- convert registers to bf16 and store into LDS ----
  auto st_tiles = [&]() {
#pragma unroll
    for (int it = 0; it < 8; ++it) {
      int slot = it * 128 + tid;
      int r = slot >> 3, c4 = (slot & 7) * 4;
      __bf16* d = &sA[r * GLDA + c4];
      d[0] = (__bf16)ra[it].x; d[1] = (__bf16)ra[it].y;
      d[2] = (__bf16)ra[it].z; d[3] = (__bf16)ra[it].w;
    }
#pragma unroll
    for (int it = 0; it < 8; ++it) {
      int slot = it * 128 + tid;
      int r = slot >> 5, c4 = (slot & 31) * 4;
      sB[(c4 + 0) * GLDA + r] = (__bf16)rb[it].x;
      sB[(c4 + 1) * GLDA + r] = (__bf16)rb[it].y;
      sB[(c4 + 2) * GLDA + r] = (__bf16)rb[it].z;
      sB[(c4 + 3) * GLDA + r] = (__bf16)rb[it].w;
    }
  };

  ld_tiles(0);
  st_tiles();

  for (int k0 = 0; k0 < K; k0 += 32) {
    __syncthreads();  // staged LDS tile visible to all waves
    const bool has_next = (k0 + 32) < K;
    if (has_next) ld_tiles(k0 + 32);  // overlap global latency with WMMAs

    bf16x16 af[4];
#pragma unroll
    for (int i = 0; i < 4; ++i)
      af[i] = ld_frag(&sA[(wm + i * 16 + row16) * GLDA + hi]);
#pragma unroll
    for (int t = 0; t < 4; ++t) {
      bf16x16 bfv = ld_frag(&sB[(wn + t * 16 + row16) * GLDA + hi]);
#pragma unroll
      for (int i = 0; i < 4; ++i) acc[i][t] = wmma_bf16(af[i], bfv, acc[i][t]);
    }
    __syncthreads();  // all waves done reading LDS
    if (has_next) st_tiles();
  }

  // ---- epilogue: C/D layout: N = lane%16, M = r + 8*(lane>=16) ----
#pragma unroll
  for (int i = 0; i < 4; ++i) {
#pragma unroll
    for (int t = 0; t < 4; ++t) {
      int nn = n0 + wn + t * 16 + row16;
      int mb = m0 + wm + i * 16 + hi;
      float bv = (epi >= 1) ? bias[nn] : 0.0f;
#pragma unroll
      for (int r = 0; r < 8; ++r) {
        float v = acc[i][t][r] + bv;
        if (epi == 2) v = fmaxf(v, 0.0f);
        C[(size_t)(mb + r) * N + nn] = v;
      }
    }
  }
}

// ---------------------------------------------------------------------------
// Flash attention: one wave (32 threads) handles one head x 16-query tile.
// DH = 64.  Key tiles of 32, online softmax, causal mask.
// ---------------------------------------------------------------------------
#define SKL 72  // sK inner stride (bf16): [key][d], 72*2=144B mult of 16
#define SVL 40  // sVt inner stride: [d][key]
#define SPL 40  // sP inner stride:  [qrow][key]

__global__ __launch_bounds__(32) void attn_kernel(
    const float* __restrict__ Q, const float* __restrict__ Km,
    const float* __restrict__ V, float* __restrict__ O, int S, int Dt) {
  __shared__ alignas(16) __bf16 sK[32 * SKL];
  __shared__ alignas(16) __bf16 sVt[64 * SVL];
  __shared__ alignas(16) __bf16 sP[16 * SPL];

  const int lane = threadIdx.x;
  const int qt = blockIdx.x, head = blockIdx.y;
  const int q0 = qt * 16;
  const int n = lane & 15;
  const int hi = (lane >> 4) << 3;
  const float scale = 0.125f;  // 1/sqrt(64)
  const int hoff = head * 64;

  // ---- Q fragments (16x64 -> two 16x32 bf16 A-frags), scaled ----
  bf16x16 aq[2];
#pragma unroll
  for (int c = 0; c < 2; ++c) {
    const float* qp = Q + (size_t)(q0 + n) * Dt + hoff + c * 32 + hi;
    float4 f0 = *(const float4*)qp;
    float4 f1 = *(const float4*)(qp + 4);
    float4 f2 = *(const float4*)(qp + 16);
    float4 f3 = *(const float4*)(qp + 20);
    float tmp[16] = {f0.x, f0.y, f0.z, f0.w, f1.x, f1.y, f1.z, f1.w,
                     f2.x, f2.y, f2.z, f2.w, f3.x, f3.y, f3.z, f3.w};
#pragma unroll
    for (int j = 0; j < 16; ++j) aq[c][j] = (__bf16)(tmp[j] * scale);
  }

  f32x8 accO[4];
#pragma unroll
  for (int t = 0; t < 4; ++t) accO[t] = (f32x8)0.0f;
  float mrun[8], lrun[8];
#pragma unroll
  for (int r = 0; r < 8; ++r) { mrun[r] = -3.0e38f; lrun[r] = 0.0f; }

  for (int kb = 0; kb <= q0 + 15; kb += 32) {
    // ---- stage K (32x64 -> [key][d]) and V (-> [d][key]) as bf16 ----
    // Load all globals into registers first (one wait), then scatter to LDS.
    int krow = kb + lane; if (krow > S - 1) krow = S - 1;
    const float* kp = Km + (size_t)krow * Dt + hoff;
    const float* vp = V + (size_t)krow * Dt + hoff;
    float4 fk[16], fv[16];
#pragma unroll
    for (int c = 0; c < 16; ++c) fk[c] = *(const float4*)(kp + c * 4);
#pragma unroll
    for (int c = 0; c < 16; ++c) fv[c] = *(const float4*)(vp + c * 4);
#pragma unroll
    for (int c = 0; c < 16; ++c) {
      __bf16* d = &sK[lane * SKL + c * 4];
      d[0] = (__bf16)fk[c].x; d[1] = (__bf16)fk[c].y;
      d[2] = (__bf16)fk[c].z; d[3] = (__bf16)fk[c].w;
      sVt[(c * 4 + 0) * SVL + lane] = (__bf16)fv[c].x;
      sVt[(c * 4 + 1) * SVL + lane] = (__bf16)fv[c].y;
      sVt[(c * 4 + 2) * SVL + lane] = (__bf16)fv[c].z;
      sVt[(c * 4 + 3) * SVL + lane] = (__bf16)fv[c].w;
    }
    __syncthreads();

    // ---- scores: S0 = Q.K^T for keys kb..kb+15, S1 for kb+16..kb+31 ----
    f32x8 s0 = (f32x8)0.0f, s1 = (f32x8)0.0f;
#pragma unroll
    for (int c = 0; c < 2; ++c) {
      bf16x16 b0 = ld_frag(&sK[n * SKL + c * 32 + hi]);
      bf16x16 b1 = ld_frag(&sK[(16 + n) * SKL + c * 32 + hi]);
      s0 = wmma_bf16(aq[c], b0, s0);
      s1 = wmma_bf16(aq[c], b1, s1);
    }

    // ---- online softmax (per row M = r + hi; cols across 16 lanes) ----
#pragma unroll
    for (int r = 0; r < 8; ++r) {
      int q = q0 + hi + r;
      float v0 = ((kb + n) <= q) ? s0[r] : -3.0e38f;
      float v1 = ((kb + 16 + n) <= q) ? s1[r] : -3.0e38f;
      float rm = hmax16(fmaxf(v0, v1));
      float mn = fmaxf(mrun[r], rm);
      float corr, p0, p1;
      if (mn <= -1.0e37f) {
        corr = 1.0f; p0 = 0.0f; p1 = 0.0f;
      } else {
        corr = __expf(mrun[r] - mn);
        p0 = __expf(v0 - mn);
        p1 = __expf(v1 - mn);
      }
      lrun[r] = lrun[r] * corr + hsum16(p0 + p1);
      mrun[r] = mn;
#pragma unroll
      for (int t = 0; t < 4; ++t) accO[t][r] *= corr;
      sP[(hi + r) * SPL + n] = (__bf16)p0;
      sP[(hi + r) * SPL + 16 + n] = (__bf16)p1;
    }
    __syncthreads();

    // ---- O += P (16x32) @ Vtile (32x64) ----
    bf16x16 ap = ld_frag(&sP[n * SPL + hi]);
#pragma unroll
    for (int t = 0; t < 4; ++t) {
      bf16x16 bv = ld_frag(&sVt[(t * 16 + n) * SVL + hi]);
      accO[t] = wmma_bf16(ap, bv, accO[t]);
    }
    __syncthreads();
  }

  // ---- write O ----
#pragma unroll
  for (int t = 0; t < 4; ++t) {
#pragma unroll
    for (int r = 0; r < 8; ++r) {
      O[(size_t)(q0 + hi + r) * Dt + hoff + t * 16 + n] = accO[t][r] / lrun[r];
    }
  }
}

// ---------------------------------------------------------------------------
// Fused (optional residual add) + LayerNorm over D=1024; one row per block.
// ---------------------------------------------------------------------------
__global__ __launch_bounds__(256) void add_ln_kernel(
    const float* __restrict__ x, const float* __restrict__ resid,
    const float* __restrict__ g, const float* __restrict__ b,
    float* __restrict__ out, int D) {
  int row = blockIdx.x;
  const float* xr = x + (size_t)row * D;
  const float* rr = resid ? resid + (size_t)row * D : nullptr;
  int tid = threadIdx.x, lane = tid & 31, wave = tid >> 5;

  float v[4];
  float sum = 0.0f, sq = 0.0f;
#pragma unroll
  for (int i = 0; i < 4; ++i) {
    int d = tid + i * 256;
    float t = xr[d] + (rr ? rr[d] : 0.0f);
    v[i] = t; sum += t; sq += t * t;
  }
#pragma unroll
  for (int off = 16; off >= 1; off >>= 1) {
    sum += __shfl_xor(sum, off, 32);
    sq  += __shfl_xor(sq, off, 32);
  }
  __shared__ float s1[8], s2[8], mb[2];
  if (lane == 0) { s1[wave] = sum; s2[wave] = sq; }
  __syncthreads();
  if (tid == 0) {
    float a = 0.0f, c = 0.0f;
#pragma unroll
    for (int w = 0; w < 8; ++w) { a += s1[w]; c += s2[w]; }
    float mean = a / (float)D;
    float var = c / (float)D - mean * mean;
    mb[0] = mean; mb[1] = rsqrtf(var + 1e-5f);
  }
  __syncthreads();
  float mean = mb[0], rstd = mb[1];
#pragma unroll
  for (int i = 0; i < 4; ++i) {
    int d = tid + i * 256;
    out[(size_t)row * D + d] = (v[i] - mean) * rstd * g[d] + b[d];
  }
}

// ---------------------------------------------------------------------------
// Host orchestration
// ---------------------------------------------------------------------------
extern "C" void kernel_launch(void* const* d_in, const int* in_sizes, int n_in,
                              void* d_out, int out_size, void* d_ws,
                              size_t ws_size, hipStream_t stream) {
  (void)in_sizes; (void)n_in; (void)out_size; (void)ws_size;
  const int S = 2048, D = 1024, H = 16, FF = 4096, L = 6;

  const int*   tokens = (const int*)d_in[0];
  const float* emb    = (const float*)d_in[1];
  const float* Wq     = (const float*)d_in[2];
  const float* Wk     = (const float*)d_in[3];
  const float* Wv     = (const float*)d_in[4];
  const float* Wo     = (const float*)d_in[5];
  const float* ln1g   = (const float*)d_in[6];
  const float* ln1b   = (const float*)d_in[7];
  const float* W1     = (const float*)d_in[8];
  const float* b1     = (const float*)d_in[9];
  const float* W2     = (const float*)d_in[10];
  const float* b2     = (const float*)d_in[11];
  const float* ln2g   = (const float*)d_in[12];
  const float* ln2b   = (const float*)d_in[13];
  const float* pg     = (const float*)d_in[14];
  const float* pb     = (const float*)d_in[15];

  float* ws = (float*)d_ws;
  const size_t SD = (size_t)S * D;
  float* h   = ws;             // persistent hidden state
  float* bQ  = ws + 1 * SD;    // Q, later reused for O
  float* bK  = ws + 2 * SD;    // K, later reused for h1
  float* bV  = ws + 3 * SD;    // V, later reused for MLP out
  float* bA  = ws + 4 * SD;    // attention output
  float* bFF = ws + 5 * SD;    // S x FF

  dim3 gemmDD(D / 128, S / 128);    // N=1024
  dim3 gemmDF(FF / 128, S / 128);   // N=4096
  dim3 attnG(S / 16, H);

  embed_kernel<<<S, 256, 0, stream>>>(tokens, emb, h, S, D);

  for (int l = 0; l < L; ++l) {
    const float* wq = Wq + (size_t)l * D * D;
    const float* wk = Wk + (size_t)l * D * D;
    const float* wv = Wv + (size_t)l * D * D;
    const float* wo = Wo + (size_t)l * D * D;
    const float* w1 = W1 + (size_t)l * D * FF;
    const float* w2 = W2 + (size_t)l * FF * D;

    gemm_kernel<<<gemmDD, 128, 0, stream>>>(h, wq, bQ, nullptr, S, D, D, 0);
    gemm_kernel<<<gemmDD, 128, 0, stream>>>(h, wk, bK, nullptr, S, D, D, 0);
    gemm_kernel<<<gemmDD, 128, 0, stream>>>(h, wv, bV, nullptr, S, D, D, 0);

    attn_kernel<<<attnG, 32, 0, stream>>>(bQ, bK, bV, bA, S, D);

    gemm_kernel<<<gemmDD, 128, 0, stream>>>(bA, wo, bQ, nullptr, S, D, D, 0);

    // h1 = LN(O + h) -> bK
    add_ln_kernel<<<S, 256, 0, stream>>>(bQ, h, ln1g + (size_t)l * D,
                                         ln1b + (size_t)l * D, bK, D);
    // ff = relu(h1 @ W1 + b1)
    gemm_kernel<<<gemmDF, 128, 0, stream>>>(bK, w1, bFF, b1 + (size_t)l * FF,
                                            S, FF, D, 2);
    // m = ff @ W2 + b2 -> bV
    gemm_kernel<<<gemmDD, 128, 0, stream>>>(bFF, w2, bV, b2 + (size_t)l * D,
                                            S, D, FF, 1);
    // h = LN(m + h1)
    add_ln_kernel<<<S, 256, 0, stream>>>(bV, bK, ln2g + (size_t)l * D,
                                         ln2b + (size_t)l * D, h, D);
  }

  add_ln_kernel<<<S, 256, 0, stream>>>(h, nullptr, pg, pb, (float*)d_out, D);
}
